// ZebrafishSNN_5k_9990093930930
// MI455X (gfx1250) — matrix operand order repaired and checked by reading, not compiled, not run on hardware
//
#include <hip/hip_runtime.h>

typedef __attribute__((ext_vector_type(16))) _Float16 v16h;
typedef __attribute__((ext_vector_type(8)))  float    v8f;
typedef __attribute__((ext_vector_type(4)))  unsigned int u4;
typedef __attribute__((ext_vector_type(4)))  float    f4;

#define TAU 0.8f
#define OMT 0.2f
#define OUT_STRIDE 700

// ---------------- packed-weight offsets (in _Float16 elements) ----------------
// Layout per tile (kt,nt): 32 lanes x 16 halves, contiguous; lane = B-fragment lane.
static const long OFF_OTL   = 0;                       // 416x608
static const long OFF_OTR   = OFF_OTL  + 416L*608;     // 416x608
static const long OFF_OTF   = OFF_OTR  + 416L*608;     // 1216x800 (two 608-row regions)
static const long OFF_PT    = OFF_OTF  + 1216L*800;    // 800x400
static const long OFF_PER   = OFF_PT   + 800L*400;     // 416x128
static const long OFF_FB    = OFF_PER  + 416L*128;     // 32x128
static const long OFF_INT   = OFF_FB   + 32L*128;      // 128x32
static const long OFF_MOTOR = OFF_INT  + 128L*32;      // 32x208
static const long OFF_CPG   = OFF_MOTOR+ 32L*208;      // 32x208
static const long OFF_EYE   = OFF_CPG  + 32L*208;      // 32x112
static const long OFF_DA    = OFF_EYE  + 32L*112;      // 32x64

// -------- pack f32 (K,N) row-major -> f16 WMMA-B tile layout, zero padded ----
__global__ void pack_w(const float* __restrict__ src, _Float16* __restrict__ dst,
                       int K, int N, int KP, int NP) {
  int idx = blockIdx.x * 256 + threadIdx.x;
  int total = KP * NP;
  if (idx >= total) return;
  int i    = idx & 15;
  int lane = (idx >> 4) & 31;
  int tile = idx >> 9;
  int ntp  = NP >> 4;
  int nt   = tile % ntp;
  int kt   = tile / ntp;
  int k = kt * 32 + ((lane & 16) >> 1) + (i & 7) + ((i & 8) << 1);     // lanes 0-15: K=0..7,16..23; lanes 16-31: K=8..15,24..31
  int n = (nt << 4) + (lane & 15);
  float v = (k < K && n < N) ? src[(long)k * N + n] : 0.f;
  dst[idx] = (_Float16)v;
}

// ---------------- fragment loaders ----------------
__device__ __forceinline__ v16h loadA_lds(const _Float16* base, int stride,
                                          int rowL, int k0, int lane) {
  int m = lane & 15, hi = lane >> 4;
  const _Float16* p = base + (rowL + m) * stride + k0 + hi * 8;
  union { v16h v; u4 q[2]; } u;
  u.q[0] = *reinterpret_cast<const u4*>(p);        // K = k0 + hi*8 .. +7
  u.q[1] = *reinterpret_cast<const u4*>(p + 16);   // K = k0 + 16 + hi*8 .. +7
  return u.v;
}

__device__ __forceinline__ v16h loadB_pk(const _Float16* wPk, int ntilesP,
                                         int kt, int nt, int lane) {
  const _Float16* p = wPk + ((((long)kt * ntilesP + nt) * 32 + lane) << 4);
  union { v16h v; u4 q[2]; } u;
  u.q[0] = *reinterpret_cast<const u4*>(p);
  u.q[1] = *reinterpret_cast<const u4*>(p + 8);
  return u.v;
}

// ---------------- one two-compartment spiking layer ----------------
__device__ __forceinline__ void snn_layer(
    const _Float16* actIn, int inStride, int ktiles,
    _Float16* actOut, int outStride,
    const _Float16* wPk, int ntilesP, int N,
    const float* vb, const float* va, const float* vs,
    int gRow0, int strip, int colgrp, int lane,
    float* outG, int outColOff,
    const _Float16* wFbPk, v16h fbFrag)
{
  int m = lane & 15, hi = lane >> 4;
  int rowL = strip * 16;
  for (int nt = colgrp; nt < ntilesP; nt += 2) {
    v8f acc = {};
    for (int kt = 0; kt < ktiles; ++kt) {
      v16h a = loadA_lds(actIn, inStride, rowL, kt * 32, lane);
      v16h b = loadB_pk(wPk, ntilesP, kt, nt, lane);
      acc = __builtin_amdgcn_wmma_f32_16x16x32_f16(false, a, false, b,
                                                   (short)0, acc, false, false);
    }
    v8f pacc = {};
    if (wFbPk) {  // apical prediction: fb @ W_fb (single K tile)
      v16h b = loadB_pk(wFbPk, ntilesP, 0, nt, lane);
      pacc = __builtin_amdgcn_wmma_f32_16x16x32_f16(false, fbFrag, false, b,
                                                    (short)0, pacc, false, false);
    }
    int n = nt * 16 + m;
    bool nv = (n < N);
#pragma unroll
    for (int r = 0; r < 8; ++r) {
      int mm = r + hi * 8;                 // C/D layout: M = r + 8*(lane>=16)
      int g  = gRow0 + rowL + mm;
      float s = 0.f;
      if (nv) {
        size_t idx = (size_t)g * N + n;
        float vbp = TAU * vb[idx] + OMT * acc[r];
        float vap = TAU * va[idx] + OMT * pacc[r];
        float vsp = TAU * vs[idx] + OMT * (vbp - vap);
        s = (vsp > 0.f) ? 1.f : 0.f;
        if (outG) outG[(size_t)g * OUT_STRIDE + outColOff + n] = s;
      }
      if (actOut) actOut[(rowL + mm) * outStride + n] = (_Float16)s;
    }
  }
}

// ---------------- affine head: intent @ W + b -> f32 out ----------------
__device__ __forceinline__ void head_layer(
    v16h aInt, const _Float16* wPk, int ntilesP, int N, const float* bias,
    int gRow0, int strip, int colgrp, int lane, float* outG, int outColOff)
{
  int m = lane & 15, hi = lane >> 4;
  int rowL = strip * 16;
  for (int nt = colgrp; nt < ntilesP; nt += 2) {
    v16h b = loadB_pk(wPk, ntilesP, 0, nt, lane);
    v8f acc = {};
    acc = __builtin_amdgcn_wmma_f32_16x16x32_f16(false, aInt, false, b,
                                                 (short)0, acc, false, false);
    int n = nt * 16 + m;
    if (n < N) {
      float bv = bias[n];
#pragma unroll
      for (int r = 0; r < 8; ++r) {
        int g = gRow0 + rowL + r + hi * 8;
        outG[(size_t)g * OUT_STRIDE + outColOff + n] = acc[r] + bv;
      }
    }
  }
}

// ---------------- feedback (spk_int) A-fragment, K=30 padded to 32 ----------
__device__ __forceinline__ v16h load_fb(const float* spk, int gRow0, int strip, int lane) {
  int m = lane & 15, hi = lane >> 4;
  const float* p = spk + (size_t)(gRow0 + strip * 16 + m) * 30;
  v16h v;
#pragma unroll
  for (int i = 0; i < 8; ++i) {
    int k1 = hi * 8 + i;        // 0..15, always valid
    int k2 = 16 + hi * 8 + i;   // 16..31, valid < 30
    v[i]     = (_Float16)p[k1];
    v[i + 8] = (_Float16)((k2 < 30) ? p[k2] : 0.f);
  }
  return v;
}

struct Params {
  const float *retina;
  const float *vb_otl, *va_otl, *vs_otl;
  const float *vb_otr, *va_otr, *vs_otr;
  const float *vb_otf, *va_otf, *vs_otf;
  const float *vb_pt,  *va_pt,  *vs_pt;
  const float *vb_per, *va_per, *vs_per;
  const float *vb_int, *va_int, *vs_int;
  const float *spk_int;
  const float *b_motor, *b_cpg, *b_eye, *b_da;
  const _Float16 *w_otl, *w_otr, *w_otf, *w_pt, *w_per, *w_fb, *w_int;
  const _Float16 *w_motor, *w_cpg, *w_eye, *w_da;
  float *out;
};

// BM=64 batch rows per block; 8 waves = 4 row strips x 2 column groups.
__global__ __launch_bounds__(256) void snn_fused(Params P) {
  __shared__ __align__(16) _Float16 bufA[64 * 832];   // retina -> otF -> per
  __shared__ __align__(16) _Float16 bufB[64 * 1216];  // [otL|otR] -> pt -> intent
  const int tid = threadIdx.x, lane = tid & 31, wid = tid >> 5;
  const int strip = wid & 3, colgrp = wid >> 2;
  const int gRow0 = blockIdx.x * 64;

  // stage retina (64 x 800 f32) -> bufA f16 (stride 832); zero pad cols 800..831
  for (int i4 = tid; i4 < 64 * 200; i4 += 256) {
    int r = i4 / 200, c4 = (i4 % 200) * 4;
    f4 f = *reinterpret_cast<const f4*>(P.retina + (size_t)(gRow0 + r) * 800 + c4);
    _Float16* d = bufA + r * 832 + c4;
    d[0] = (_Float16)f[0]; d[1] = (_Float16)f[1];
    d[2] = (_Float16)f[2]; d[3] = (_Float16)f[3];
  }
  for (int i = tid; i < 64 * 32; i += 256)
    bufA[(i / 32) * 832 + 800 + (i % 32)] = (_Float16)0.f;
  __syncthreads();

  v16h zf = {};

  // otL: retina[:, :400] @ W_otl -> bufB[:, 0:608)     (K 400->13kt, N 600->38nt)
  snn_layer(bufA,       832, 13, bufB,       1216, P.w_otl, 38, 600,
            P.vb_otl, P.va_otl, P.vs_otl, gRow0, strip, colgrp, lane,
            nullptr, 0, nullptr, zf);
  // otR: retina[:, 400:800] @ W_otr -> bufB[:, 608:1216)
  snn_layer(bufA + 400, 832, 13, bufB + 608, 1216, P.w_otr, 38, 600,
            P.vb_otr, P.va_otr, P.vs_otr, gRow0, strip, colgrp, lane,
            nullptr, 0, nullptr, zf);
  __syncthreads();

  // otF: fused(1216p) @ W_otf -> bufA[:, 0:800)        (38kt, 50nt)
  snn_layer(bufB, 1216, 38, bufA, 832, P.w_otf, 50, 800,
            P.vb_otf, P.va_otf, P.vs_otf, gRow0, strip, colgrp, lane,
            nullptr, 0, nullptr, zf);
  __syncthreads();

  // pt: otF(800) @ W_pt -> bufB[:, 0:400)              (25kt, 25nt)
  snn_layer(bufA, 832, 25, bufB, 1216, P.w_pt, 25, 400,
            P.vb_pt, P.va_pt, P.vs_pt, gRow0, strip, colgrp, lane,
            nullptr, 0, nullptr, zf);
  __syncthreads();

  // per: pt(400->416p) @ W_per, pred = spk_int @ W_fb -> bufA[:, 0:128) + out[550..669]
  v16h fbFrag = load_fb(P.spk_int, gRow0, strip, lane);
  snn_layer(bufB, 1216, 13, bufA, 832, P.w_per, 8, 120,
            P.vb_per, P.va_per, P.vs_per, gRow0, strip, colgrp, lane,
            P.out, 550, P.w_fb, fbFrag);
  __syncthreads();

  // intent: per(120->128p) @ W_int -> bufB[:, 0:32) + out[670..699]
  snn_layer(bufA, 832, 4, bufB, 1216, P.w_int, 2, 30,
            P.vb_int, P.va_int, P.vs_int, gRow0, strip, colgrp, lane,
            P.out, 670, nullptr, zf);
  __syncthreads();

  // affine heads from intent (single K-tile of 32, cols 30/31 are zero)
  v16h aInt = loadA_lds(bufB, 1216, strip * 16, 0, lane);
  head_layer(aInt, P.w_motor, 13, 200, P.b_motor, gRow0, strip, colgrp, lane, P.out,   0);
  head_layer(aInt, P.w_cpg,   13, 200, P.b_cpg,   gRow0, strip, colgrp, lane, P.out, 200);
  head_layer(aInt, P.w_eye,    7, 100, P.b_eye,   gRow0, strip, colgrp, lane, P.out, 400);
  head_layer(aInt, P.w_da,     4,  50, P.b_da,    gRow0, strip, colgrp, lane, P.out, 500);
}

extern "C" void kernel_launch(void* const* d_in, const int* in_sizes, int n_in,
                              void* d_out, int out_size, void* d_ws, size_t ws_size,
                              hipStream_t stream) {
  (void)in_sizes; (void)n_in; (void)out_size; (void)ws_size;
  _Float16* ws = (_Float16*)d_ws;

  auto pack = [&](const void* src, long srcRowOff, long dstOff,
                  int K, int N, int KP, int NP) {
    int total = KP * NP;
    pack_w<<<(total + 255) / 256, 256, 0, stream>>>(
        (const float*)src + srcRowOff, ws + dstOff, K, N, KP, NP);
  };
  pack(d_in[1],  0,        OFF_OTL,           400, 600, 416, 608);
  pack(d_in[2],  0,        OFF_OTR,           400, 600, 416, 608);
  pack(d_in[3],  0,        OFF_OTF,           600, 800, 608, 800); // W_otf rows 0..599 (otL)
  pack(d_in[3],  600L*800, OFF_OTF + 608L*800,600, 800, 608, 800); // W_otf rows 600..1199 (otR)
  pack(d_in[4],  0,        OFF_PT,            800, 400, 800, 400);
  pack(d_in[5],  0,        OFF_PER,           400, 120, 416, 128);
  pack(d_in[6],  0,        OFF_FB,             30, 120,  32, 128);
  pack(d_in[7],  0,        OFF_INT,           120,  30, 128,  32);
  pack(d_in[8],  0,        OFF_MOTOR,          30, 200,  32, 208);
  pack(d_in[10], 0,        OFF_CPG,            30, 200,  32, 208);
  pack(d_in[12], 0,        OFF_EYE,            30, 100,  32, 112);
  pack(d_in[14], 0,        OFF_DA,             30,  50,  32,  64);

  Params P;
  P.retina = (const float*)d_in[0];
  P.vb_otl = (const float*)d_in[16]; P.va_otl = (const float*)d_in[17]; P.vs_otl = (const float*)d_in[18];
  P.vb_otr = (const float*)d_in[19]; P.va_otr = (const float*)d_in[20]; P.vs_otr = (const float*)d_in[21];
  P.vb_otf = (const float*)d_in[22]; P.va_otf = (const float*)d_in[23]; P.vs_otf = (const float*)d_in[24];
  P.vb_pt  = (const float*)d_in[25]; P.va_pt  = (const float*)d_in[26]; P.vs_pt  = (const float*)d_in[27];
  P.vb_per = (const float*)d_in[28]; P.va_per = (const float*)d_in[29]; P.vs_per = (const float*)d_in[30];
  P.vb_int = (const float*)d_in[31]; P.va_int = (const float*)d_in[32]; P.vs_int = (const float*)d_in[33];
  P.spk_int = (const float*)d_in[34];
  P.b_motor = (const float*)d_in[9];  P.b_cpg = (const float*)d_in[11];
  P.b_eye   = (const float*)d_in[13]; P.b_da  = (const float*)d_in[15];
  P.w_otl = ws + OFF_OTL; P.w_otr = ws + OFF_OTR; P.w_otf = ws + OFF_OTF;
  P.w_pt  = ws + OFF_PT;  P.w_per = ws + OFF_PER; P.w_fb  = ws + OFF_FB;
  P.w_int = ws + OFF_INT;
  P.w_motor = ws + OFF_MOTOR; P.w_cpg = ws + OFF_CPG;
  P.w_eye   = ws + OFF_EYE;   P.w_da  = ws + OFF_DA;
  P.out = (float*)d_out;

  snn_fused<<<16384 / 64, 256, 0, stream>>>(P);
}
